// MixedChunkAttention_36404142801557
// MI455X (gfx1250) — matrix operand order repaired and checked by reading, not compiled, not run on hardware
//
#include <hip/hip_runtime.h>
#include <math.h>

// ---------------------------------------------------------------------------
// MixedChunkAttention (FLASH/GAU-style) for MI455X (gfx1250), wave32 + WMMA.
// GEMM core: V_WMMA_F32_16X16X32_BF16 with ping-pong LDS buffers.
// Row-major staging uses CDNA5 GLOBAL_LOAD_ASYNC_TO_LDS_B128 (ASYNCcnt);
// convert/transpose staging is register-pipelined (commit after WMMA).
// ---------------------------------------------------------------------------

typedef __bf16 bf16;
typedef __attribute__((ext_vector_type(16))) __bf16 bf16x16;
typedef __attribute__((ext_vector_type(8)))  __bf16 bf16x8;
typedef __attribute__((ext_vector_type(8)))  float  f32x8;

#define NTHREADS 256      // 8 waves of 32
#define TBM 128           // block tile M
#define TBN 128           // block tile N
#define TBK 32            // K step = WMMA K
#define LDSK 40           // padded row length (80B stride: 16B aligned, conflict-free-ish)
#define TILE_ELEMS (TBM * LDSK)

// Problem constants
#define BATCH 4
#define SEQ   4096
#define DIM   1024
#define HID   2048        // H
#define HID2  4096        // 2H
#define QD    128
#define GRP   256         // chunk size G
#define NCHUNK 64         // BATCH * SEQ/GRP
#define ROWS  16384       // BATCH*SEQ

enum { S_BF16_RM = 0, S_F32_RM = 1, S_BF16_TR = 2 };

struct Acc { f32x8 a[4][2]; };

__device__ __forceinline__ void acc_zero(Acc& A) {
#pragma unroll
  for (int i = 0; i < 4; ++i)
#pragma unroll
    for (int j = 0; j < 2; ++j)
#pragma unroll
      for (int e = 0; e < 8; ++e) A.a[i][j][e] = 0.0f;
}

// fast silu: x * v_rcp_f32(1 + exp(-x))  (avoids the IEEE div expansion)
__device__ __forceinline__ float silu_f(float x) {
  return x * __builtin_amdgcn_rcpf(1.0f + __expf(-x));
}

// --- CDNA5 async memory->LDS copy (16B per lane), tracked by ASYNCcnt --------
// VDST = 32-bit LDS byte offset (low 32 bits of a generic __shared__ pointer),
// VADDR = 64-bit global address.
__device__ __forceinline__ void async_copy_b128(void* lds, const void* gaddr) {
  asm volatile("global_load_async_to_lds_b128 %0, %1, off"
               :: "v"((unsigned)(uintptr_t)lds), "v"(gaddr)
               : "memory");
}
__device__ __forceinline__ void async_wait0() {
  asm volatile("s_wait_asynccnt 0" ::: "memory");
}

// --- staging of a 128 x 32 tile into LDS (layout [r][k], row stride LDSK) ----
// S_BF16_RM : lds[r][k] = src[(r0+r)*ld + k0+k]   -- async DMA, no registers
// S_F32_RM  : same with f32->bf16 convert         -- issue loads / commit stores
// S_BF16_TR : lds[r][k] = src[(k0+k)*ld + r0+r]   -- issue loads / commit scatter
struct StageBuf { float4 f[4]; bf16x8 h[2]; };

template<int MODE>
__device__ __forceinline__ void stage_issue(StageBuf& rb, bf16* lds, const void* src,
                                            int ld, int r0, int k0, int tid) {
  if constexpr (MODE == S_BF16_RM) {
#pragma unroll
    for (int u = 0; u < 2; ++u) {
      int idx = tid + u * NTHREADS;          // 512 units of 8 bf16 (16B)
      int r = idx >> 2, seg = idx & 3;
      const bf16* gp = (const bf16*)src + (size_t)(r0 + r) * ld + k0 + seg * 8;
      async_copy_b128(lds + r * LDSK + seg * 8, gp);
    }
  } else if constexpr (MODE == S_F32_RM) {
#pragma unroll
    for (int u = 0; u < 2; ++u) {
      int idx = tid + u * NTHREADS;
      int r = idx >> 2, seg = idx & 3;
      const float* gp = (const float*)src + (size_t)(r0 + r) * ld + k0 + seg * 8;
      rb.f[2 * u + 0] = *(const float4*)gp;
      rb.f[2 * u + 1] = *(const float4*)(gp + 4);
    }
  } else {  // S_BF16_TR: coalesced 16B reads along fast axis
#pragma unroll
    for (int u = 0; u < 2; ++u) {
      int idx = tid + u * NTHREADS;          // 512 units: 32 k x 16 row-segments
      int k = idx >> 4, nseg = idx & 15;
      const bf16* gp = (const bf16*)src + (size_t)(k0 + k) * ld + r0 + nseg * 8;
      if (u == 0) __builtin_prefetch((const char*)gp + 128, 0, 0);
      rb.h[u] = *(const bf16x8*)gp;
    }
  }
}

template<int MODE>
__device__ __forceinline__ void stage_commit(const StageBuf& rb, bf16* lds, int tid) {
  if constexpr (MODE == S_BF16_RM) {
    // async path: data lands in LDS directly; nothing to do here
  } else if constexpr (MODE == S_F32_RM) {
#pragma unroll
    for (int u = 0; u < 2; ++u) {
      int idx = tid + u * NTHREADS;
      int r = idx >> 2, seg = idx & 3;
      float4 v0 = rb.f[2 * u + 0], v1 = rb.f[2 * u + 1];
      bf16x8 o;
      o[0] = (bf16)v0.x; o[1] = (bf16)v0.y; o[2] = (bf16)v0.z; o[3] = (bf16)v0.w;
      o[4] = (bf16)v1.x; o[5] = (bf16)v1.y; o[6] = (bf16)v1.z; o[7] = (bf16)v1.w;
      *(bf16x8*)(lds + r * LDSK + seg * 8) = o;
    }
  } else {
#pragma unroll
    for (int u = 0; u < 2; ++u) {
      int idx = tid + u * NTHREADS;
      int k = idx >> 4, nseg = idx & 15;
#pragma unroll
      for (int e = 0; e < 8; ++e) lds[(nseg * 8 + e) * LDSK + k] = rb.h[u][e];
    }
  }
}

template<int AMODE, int BMODE>
__device__ __forceinline__ void maybe_async_wait() {
  if constexpr (AMODE == S_BF16_RM || BMODE == S_BF16_RM) async_wait0();
}

// --- shared WMMA GEMM inner loop (ping-pong, software pipelined) -------------
// acc += A[128 x K] * B[K x 128]
// 8 waves: wm = wave&1 (64 rows each), wn = wave>>1 (32 cols each).
// ISA bf16 A 16x32 frag: lane<16 -> row=lane,    K {0..7,16..23}
//                        lane>=16 -> row=lane-16, K {8..15,24..31}
// ISA bf16 B 32x16 frag: lane<16 -> col=lane, K {0..7,16..23} ; lane>=16 -> col, K {8..15,24..31}
template<int AMODE, int BMODE>
__device__ __forceinline__ void gemm_loop(Acc& acc,
    const void* A, int lda, int r0A, int k0A,
    const void* B, int ldb, int r0B, int k0B, int K,
    bf16* sA, bf16* sB) {          // each holds 2 tiles (ping-pong)
  const int tid  = threadIdx.x;
  const int lane = tid & 31;
  const int wave = tid >> 5;
  const int wm   = wave & 1;
  const int wn   = wave >> 1;
  const int half = lane >> 4;
  const int l15  = lane & 15;

  StageBuf ra, rb;

  // prologue: stage tile 0 into buffer 0
  stage_issue<AMODE>(ra, sA, A, lda, r0A, k0A, tid);
  stage_issue<BMODE>(rb, sB, B, ldb, r0B, k0B, tid);
  stage_commit<AMODE>(ra, sA, tid);
  stage_commit<BMODE>(rb, sB, tid);
  maybe_async_wait<AMODE, BMODE>();
  __syncthreads();

  for (int kk = 0; kk < K; kk += TBK) {
    const int cur = (kk / TBK) & 1;
    bf16* cA = sA + cur * TILE_ELEMS;
    bf16* cB = sB + cur * TILE_ELEMS;
    const bool has_next = (kk + TBK) < K;
    bf16* nA = sA + (cur ^ 1) * TILE_ELEMS;
    bf16* nB = sB + (cur ^ 1) * TILE_ELEMS;

    // 1) issue next tile's loads (async DMA straight to LDS, or into registers)
    if (has_next) {
      stage_issue<AMODE>(ra, nA, A, lda, r0A, k0A + kk + TBK, tid);
      stage_issue<BMODE>(rb, nB, B, ldb, r0B, k0B + kk + TBK, tid);
    }

    // 2) consume current tile: fragment loads + 8 WMMAs
    bf16x16 af[4], bfr[2];
#pragma unroll
    for (int tm = 0; tm < 4; ++tm) {
      const bf16* p = cA + (wm * 64 + tm * 16 + l15) * LDSK;
      bf16x8 lo = *(const bf16x8*)(p + half * 8);
      bf16x8 hi = *(const bf16x8*)(p + 16 + half * 8);
      af[tm] = __builtin_shufflevector(lo, hi, 0,1,2,3,4,5,6,7,8,9,10,11,12,13,14,15);
    }
#pragma unroll
    for (int tn = 0; tn < 2; ++tn) {
      const bf16* p = cB + (wn * 32 + tn * 16 + l15) * LDSK;
      bf16x8 lo = *(const bf16x8*)(p + half * 8);
      bf16x8 hi = *(const bf16x8*)(p + 16 + half * 8);
      bfr[tn] = __builtin_shufflevector(lo, hi, 0,1,2,3,4,5,6,7,8,9,10,11,12,13,14,15);
    }
#pragma unroll
    for (int tm = 0; tm < 4; ++tm)
#pragma unroll
      for (int tn = 0; tn < 2; ++tn)
        acc.a[tm][tn] = __builtin_amdgcn_wmma_f32_16x16x32_bf16(
            false, af[tm], false, bfr[tn], (short)0, acc.a[tm][tn], false, false);

    // 3) commit next tile to LDS (register modes wait for loads only here,
    //    i.e. after the WMMA stream), then make LDS writes visible
    if (has_next) {
      stage_commit<AMODE>(ra, nA, tid);
      stage_commit<BMODE>(rb, nB, tid);
      maybe_async_wait<AMODE, BMODE>();
    }
    __syncthreads();
  }
}

// --- epilogue walker: C frag layout -> (row, col, value) ---------------------
// VGPR r, lane L: row = 16*tm + r + (L>=16 ? 8 : 0), col = 16*tn + (L&15)
// colfn(col) builds a per-column context once (each thread owns 2 columns);
// f(row, col, val, ctx) handles one element.
template<typename FC, typename FE>
__device__ __forceinline__ void epi_apply(const Acc& acc, int m0, int n0,
                                          FC colfn, FE f) {
  const int tid  = threadIdx.x;
  const int lane = tid & 31;
  const int wave = tid >> 5;
  const int wm = wave & 1, wn = wave >> 1;
  const int half = lane >> 4, l15 = lane & 15;
#pragma unroll
  for (int tn = 0; tn < 2; ++tn) {
    const int col = n0 + wn * 32 + tn * 16 + l15;
    const auto ctx = colfn(col);
#pragma unroll
    for (int tm = 0; tm < 4; ++tm)
#pragma unroll
      for (int r = 0; r < 8; ++r) {
        int row = m0 + wm * 64 + tm * 16 + r + half * 8;
        f(row, col, acc.a[tm][tn][r], ctx);
      }
  }
}

#define GEMM_SHARED() \
  __shared__ __align__(16) bf16 sA[2 * TILE_ELEMS]; \
  __shared__ __align__(16) bf16 sB[2 * TILE_ELEMS];

// ============================ small helper kernels ===========================

__global__ void k_f32_to_bf16(const float* __restrict__ src, bf16* __restrict__ dst, int n) {
  for (int i = blockIdx.x * blockDim.x + threadIdx.x; i < n; i += gridDim.x * blockDim.x)
    dst[i] = (bf16)src[i];
}

// T5 relative position bias table [G, G], already scaled by sqrt(128)
__global__ void k_bias(const float* __restrict__ rel_emb, float* __restrict__ bias) {
  int i = blockIdx.x;        // query pos
  int j = threadIdx.x;       // key pos
  int n = i - j;             // n = -(k_pos - q_pos)
  int ret = (n < 0) ? 16 : 0;
  n = abs(n);
  int bucket;
  if (n < 8) {
    bucket = ret + n;
  } else {
    int vl = 8 + (int)(logf((float)n * 0.125f) * (1.0f / logf(16.0f)) * 8.0f);
    bucket = ret + min(vl, 15);
  }
  bias[i * GRP + j] = rel_emb[bucket] * 11.313708498984761f;  // * sqrt(128)
}

// LayerNorm over D=1024; one block (256 thr) per row; bf16 output
__global__ __launch_bounds__(NTHREADS) void k_layernorm(
    const float* __restrict__ v, const float* __restrict__ g,
    const float* __restrict__ b, bf16* __restrict__ out) {
  __shared__ float red[NTHREADS];
  const int row = blockIdx.x, tid = threadIdx.x;
  const float4 x = *(const float4*)(v + (size_t)row * DIM + tid * 4);
  float s  = x.x + x.y + x.z + x.w;
  float sq = x.x * x.x + x.y * x.y + x.z * x.z + x.w * x.w;

  red[tid] = s; __syncthreads();
  for (int off = 128; off; off >>= 1) { if (tid < off) red[tid] += red[tid + off]; __syncthreads(); }
  const float mean = red[0] * (1.0f / DIM);
  __syncthreads();
  red[tid] = sq; __syncthreads();
  for (int off = 128; off; off >>= 1) { if (tid < off) red[tid] += red[tid + off]; __syncthreads(); }
  const float var  = red[0] * (1.0f / DIM) - mean * mean;
  const float rinv = __builtin_amdgcn_rsqf(var + 1e-5f);

  const int c = tid * 4;
  bf16* o = out + (size_t)row * DIM + c;
  o[0] = (bf16)((x.x - mean) * rinv * g[c + 0] + b[c + 0]);
  o[1] = (bf16)((x.y - mean) * rinv * g[c + 1] + b[c + 1]);
  o[2] = (bf16)((x.z - mean) * rinv * g[c + 2] + b[c + 2]);
  o[3] = (bf16)((x.w - mean) * rinv * g[c + 3] + b[c + 3]);
}

// ============================== GEMM kernels ================================

// hid = silu(normed @ w_hid + b_hid); split into vv / gate (bf16)
__global__ __launch_bounds__(NTHREADS) void k_gemm_hid(
    const bf16* __restrict__ A, const bf16* __restrict__ W,
    const float* __restrict__ bvec, bf16* __restrict__ vv, bf16* __restrict__ gate) {
  GEMM_SHARED();
  const int m0 = blockIdx.x * TBM, n0 = blockIdx.y * TBN;
  Acc acc; acc_zero(acc);
  gemm_loop<S_BF16_RM, S_BF16_TR>(acc, A, DIM, m0, 0, W, HID2, n0, 0, DIM, sA, sB);
  epi_apply(acc, m0, n0,
    [&](int c) { return bvec[c]; },
    [&](int r, int c, float val, float bv) {
      float s = silu_f(val + bv);
      if (c < HID) vv[(size_t)r * HID + c] = (bf16)s;
      else         gate[(size_t)r * HID + (c - HID)] = (bf16)s;
    });
}

struct ProjCtx { float bv, g0, b0, g1, b1; };

// proj = silu(X @ W + b); quad = proj*g0+b0, lin = proj*g1+b1 (bf16)
__global__ __launch_bounds__(NTHREADS) void k_gemm_proj(
    const float* __restrict__ X, const bf16* __restrict__ W,
    const float* __restrict__ bvec, const float* __restrict__ gam,
    const float* __restrict__ bet, bf16* __restrict__ quad, bf16* __restrict__ lin) {
  GEMM_SHARED();
  const int m0 = blockIdx.x * TBM;
  Acc acc; acc_zero(acc);
  gemm_loop<S_F32_RM, S_BF16_TR>(acc, X, DIM, m0, 0, W, QD, 0, 0, DIM, sA, sB);
  epi_apply(acc, m0, 0,
    [&](int c) { return ProjCtx{bvec[c], gam[c], bet[c], gam[QD + c], bet[QD + c]}; },
    [&](int r, int c, float val, const ProjCtx& cx) {
      float s = silu_f(val + cx.bv);
      quad[(size_t)r * QD + c] = (bf16)(s * cx.g0 + cx.b0);
      lin [(size_t)r * QD + c] = (bf16)(s * cx.g1 + cx.b1);
    });
}

// per-chunk: attn = relu(qq@qk^T/G + bias)^2 ; fp32 -> d_out, bf16 -> ws
__global__ __launch_bounds__(NTHREADS) void k_gemm_sim(
    const bf16* __restrict__ qq, const bf16* __restrict__ qk,
    const float* __restrict__ bias, float* __restrict__ attn_out,
    bf16* __restrict__ attn_bf) {
  GEMM_SHARED();
  const int chunk = blockIdx.z;
  const int m0 = blockIdx.x * TBM, n0 = blockIdx.y * TBN;   // within-chunk
  Acc acc; acc_zero(acc);
  // sim[i][j] = sum_d qq[i][d]*qk[j][d]  -> B is row-major [N,K] = qk rows
  gemm_loop<S_BF16_RM, S_BF16_RM>(acc, qq, QD, chunk * GRP + m0, 0,
                                  qk, QD, chunk * GRP + n0, 0, QD, sA, sB);
  epi_apply(acc, m0, n0,
    [&](int j) { return j; },
    [&](int i, int j, float val, int) {
      float s = val * (1.0f / GRP) + bias[i * GRP + j];
      float r = fmaxf(s, 0.0f);
      float a = r * r;
      attn_out[(size_t)(chunk * GRP + i) * GRP + j] = a;
      attn_bf[(size_t)chunk * GRP * GRP + i * GRP + j] = (bf16)a;
    });
}

// lin_kv[b] = (lin_k[b]^T @ vv[b]) / SEQ   (128 x 2048 per batch, K = 4096)
__global__ __launch_bounds__(NTHREADS) void k_gemm_linkv(
    const bf16* __restrict__ lk, const bf16* __restrict__ vv, bf16* __restrict__ linkv) {
  GEMM_SHARED();
  const int b = blockIdx.z, n0 = blockIdx.y * TBN;
  Acc acc; acc_zero(acc);
  gemm_loop<S_BF16_TR, S_BF16_TR>(acc, lk, QD, 0, b * SEQ,
                                  vv, HID, n0, b * SEQ, SEQ, sA, sB);
  epi_apply(acc, 0, n0,
    [&](int e) { return e; },
    [&](int d, int e, float val, int) {
      linkv[(size_t)b * QD * HID + (size_t)d * HID + e] = (bf16)(val * (1.0f / SEQ));
    });
}

// mix = gate * (attn @ cv + lin_q @ lin_kv)   (two fused accumulation phases)
__global__ __launch_bounds__(NTHREADS) void k_gemm_mix(
    const bf16* __restrict__ attnbf, const bf16* __restrict__ vv,
    const bf16* __restrict__ lq, const bf16* __restrict__ linkv,
    const bf16* __restrict__ gate, bf16* __restrict__ mix) {
  GEMM_SHARED();
  const int chunk = blockIdx.z;
  const int b = chunk >> 4;
  const int m0c = blockIdx.x * TBM;       // within chunk (0 or 128)
  const int n0 = blockIdx.y * TBN;
  Acc acc; acc_zero(acc);
  // phase 1: quad_out = attn(256x256) @ cv(256x2048)
  gemm_loop<S_BF16_RM, S_BF16_TR>(acc, attnbf + (size_t)chunk * GRP * GRP, GRP, m0c, 0,
                                  vv, HID, n0, chunk * GRP, GRP, sA, sB);
  // phase 2: lin_out = lin_q(256x128) @ lin_kv[b](128x2048)
  gemm_loop<S_BF16_RM, S_BF16_TR>(acc, lq, QD, chunk * GRP + m0c, 0,
                                  linkv + (size_t)b * QD * HID, HID, n0, 0, QD, sA, sB);
  epi_apply(acc, chunk * GRP + m0c, n0,
    [&](int c) { return c; },
    [&](int r, int c, float val, int) {
      float g = (float)gate[(size_t)r * HID + c];
      mix[(size_t)r * HID + c] = (bf16)(g * val);
    });
}

// out = mix @ w_out + b_out   (fp32 result)
__global__ __launch_bounds__(NTHREADS) void k_gemm_out(
    const bf16* __restrict__ mix, const bf16* __restrict__ W,
    const float* __restrict__ bvec, float* __restrict__ out) {
  GEMM_SHARED();
  const int m0 = blockIdx.x * TBM, n0 = blockIdx.y * TBN;
  Acc acc; acc_zero(acc);
  gemm_loop<S_BF16_RM, S_BF16_TR>(acc, mix, HID, m0, 0, W, DIM, n0, 0, HID, sA, sB);
  epi_apply(acc, m0, n0,
    [&](int c) { return bvec[c]; },
    [&](int r, int c, float val, float bv) {
      out[(size_t)r * DIM + c] = val + bv;
    });
}

// ================================ launcher ==================================

extern "C" void kernel_launch(void* const* d_in, const int* in_sizes, int n_in,
                              void* d_out, int out_size, void* d_ws, size_t ws_size,
                              hipStream_t stream) {
  (void)in_sizes; (void)n_in; (void)out_size; (void)ws_size;
  const float* q     = (const float*)d_in[0];
  const float* k     = (const float*)d_in[1];
  const float* v     = (const float*)d_in[2];
  const float* ln_g  = (const float*)d_in[3];
  const float* ln_b  = (const float*)d_in[4];
  const float* w_hid = (const float*)d_in[5];
  const float* b_hid = (const float*)d_in[6];
  const float* w_q   = (const float*)d_in[7];
  const float* b_q   = (const float*)d_in[8];
  const float* w_k   = (const float*)d_in[9];
  const float* b_k   = (const float*)d_in[10];
  const float* qs_g  = (const float*)d_in[11];
  const float* qs_b  = (const float*)d_in[12];
  const float* ks_g  = (const float*)d_in[13];
  const float* ks_b  = (const float*)d_in[14];
  const float* rel   = (const float*)d_in[15];
  const float* w_out = (const float*)d_in[16];
  const float* b_out = (const float*)d_in[17];

  // ---- carve workspace (all 256B aligned) ----
  char* wsp = (char*)d_ws;
  auto carve = [&](size_t bytes) -> char* {
    char* p = wsp; wsp += (bytes + 255) & ~(size_t)255; return p;
  };
  float* bias   = (float*)carve((size_t)GRP * GRP * 4);
  bf16* normed  = (bf16*)carve((size_t)ROWS * DIM * 2);
  bf16* whid_b  = (bf16*)carve((size_t)DIM * HID2 * 2);
  bf16* wq_b    = (bf16*)carve((size_t)DIM * QD * 2);
  bf16* wk_b    = (bf16*)carve((size_t)DIM * QD * 2);
  bf16* wout_b  = (bf16*)carve((size_t)HID * DIM * 2);
  bf16* vv      = (bf16*)carve((size_t)ROWS * HID * 2);
  bf16* gate    = (bf16*)carve((size_t)ROWS * HID * 2);
  bf16* qq      = (bf16*)carve((size_t)ROWS * QD * 2);
  bf16* lq      = (bf16*)carve((size_t)ROWS * QD * 2);
  bf16* qk      = (bf16*)carve((size_t)ROWS * QD * 2);
  bf16* lk      = (bf16*)carve((size_t)ROWS * QD * 2);
  bf16* attnbf  = (bf16*)carve((size_t)NCHUNK * GRP * GRP * 2);
  bf16* linkv   = (bf16*)carve((size_t)BATCH * QD * HID * 2);
  bf16* mixb    = (bf16*)carve((size_t)ROWS * HID * 2);

  float* out      = (float*)d_out;                       // [4,4096,1024]
  float* attn_out = out + (size_t)BATCH * SEQ * DIM;     // [4,4096,256]

  // ---- stage 0: weight conversion + bias table + layernorm ----
  k_f32_to_bf16<<<1024, NTHREADS, 0, stream>>>(w_hid, whid_b, DIM * HID2);
  k_f32_to_bf16<<<128,  NTHREADS, 0, stream>>>(w_q,   wq_b,   DIM * QD);
  k_f32_to_bf16<<<128,  NTHREADS, 0, stream>>>(w_k,   wk_b,   DIM * QD);
  k_f32_to_bf16<<<1024, NTHREADS, 0, stream>>>(w_out, wout_b, HID * DIM);
  k_bias<<<GRP, GRP, 0, stream>>>(rel, bias);
  k_layernorm<<<ROWS, NTHREADS, 0, stream>>>(v, ln_g, ln_b, normed);

  // ---- stage 1: big projections ----
  k_gemm_hid<<<dim3(ROWS / TBM, HID2 / TBN, 1), NTHREADS, 0, stream>>>(
      normed, whid_b, b_hid, vv, gate);
  k_gemm_proj<<<dim3(ROWS / TBM, 1, 1), NTHREADS, 0, stream>>>(
      q, wq_b, b_q, qs_g, qs_b, qq, lq);
  k_gemm_proj<<<dim3(ROWS / TBM, 1, 1), NTHREADS, 0, stream>>>(
      k, wk_b, b_k, ks_g, ks_b, qk, lk);

  // ---- stage 2: quadratic attention scores + linear-attention KV ----
  k_gemm_sim<<<dim3(GRP / TBM, GRP / TBN, NCHUNK), NTHREADS, 0, stream>>>(
      qq, qk, bias, attn_out, attnbf);
  k_gemm_linkv<<<dim3(1, HID / TBN, BATCH), NTHREADS, 0, stream>>>(lk, vv, linkv);

  // ---- stage 3: fused quad_out + lin_out, gated ----
  k_gemm_mix<<<dim3(GRP / TBM, HID / TBN, NCHUNK), NTHREADS, 0, stream>>>(
      attnbf, vv, lq, linkv, gate, mixb);

  // ---- stage 4: output projection ----
  k_gemm_out<<<dim3(ROWS / TBM, DIM / TBN, 1), NTHREADS, 0, stream>>>(
      mixb, wout_b, b_out, out);
}
